// SelectSphereConv_31731218383018
// MI455X (gfx1250) — matrix operand order repaired and checked by reading, not compiled, not run on hardware
//
#include <hip/hip_runtime.h>

#define Bc 4
#define Cc 64
#define Vc 40962
#define Kc 9
#define COUTc 64
#define CK 576          // C*K contraction length = 18 * 32
#define PADK 584        // padded LDS row stride (conflict-free: 584*2/4 = 292 words, 292%64=36)
#define VT 16           // vertices per block
#define NKSTEP 18       // 576 / 32

typedef __attribute__((ext_vector_type(16))) __bf16 v16bf;
typedef __attribute__((ext_vector_type(8)))  __bf16 v8bf;
typedef __attribute__((ext_vector_type(8)))  float  v8f;

// LDS byte offsets inside one dynamic-shared allocation
#define OFF_W    0          // 64 rows * 584 bf16 = 74752 B
#define OFF_ITP  74752      // 64 cols * 584 bf16 = 74752 B
#define OFF_MAT  149504     // 16 * 81 f32 = 5184 B
#define OFF_IDX  154688     // 16 * 9 i32 = 576 B
#define SMEM_BYTES 155264

static __device__ __forceinline__ unsigned short f2bf(float f) {
    unsigned int u = __builtin_bit_cast(unsigned int, f);
    u += 0x7FFFu + ((u >> 16) & 1u);       // round-to-nearest-even
    return (unsigned short)(u >> 16);
}

// ---------------------------------------------------------------------------
// Pass 1: transpose x [B*C=256][V] -> xT [V][256] so neighbor gathers coalesce
// ---------------------------------------------------------------------------
__global__ void __launch_bounds__(256) transpose_x_kernel(
        const float* __restrict__ x, float* __restrict__ xT) {
    __shared__ float t[64][65];
    const int v0 = blockIdx.x * 64;
    const int p0 = blockIdx.y * 64;          // p = b*C + c, gridDim.y = 4
    const int tx = threadIdx.x & 63;
    const int ty = threadIdx.x >> 6;         // 0..3
    #pragma unroll
    for (int r = 0; r < 16; ++r) {
        int p = p0 + ty * 16 + r;
        int v = v0 + tx;
        t[ty * 16 + r][tx] = (v < Vc) ? x[(size_t)p * Vc + v] : 0.0f;
    }
    __syncthreads();
    #pragma unroll
    for (int r = 0; r < 16; ++r) {
        int v = v0 + ty * 16 + r;
        if (v < Vc) xT[(size_t)v * 256 + p0 + tx] = t[tx][ty * 16 + r];
    }
}

// ---------------------------------------------------------------------------
// Pass 2: fused gather + 9x9 interp (f32 VALU) + conv GEMM (bf16 WMMA)
// ---------------------------------------------------------------------------
__global__ void __launch_bounds__(256) sphere_conv_fused_kernel(
        const float* __restrict__ x,        // [B,C,V] (fallback path)
        const float* __restrict__ xT,       // [V,256] (fast path)
        int useT,
        const int*   __restrict__ index,    // [V,9]
        const float* __restrict__ itp_mat,  // [V,9,9]
        const float* __restrict__ conv_w,   // [64,64,1,9] = [64][576]
        const float* __restrict__ conv_b,   // [64]
        float*       __restrict__ out) {    // [B,64,V]
    extern __shared__ char smem[];
    unsigned short* sW   = (unsigned short*)(smem + OFF_W);
    unsigned short* sITP = (unsigned short*)(smem + OFF_ITP);
    float*          sMat = (float*)(smem + OFF_MAT);
    int*            sIdx = (int*)(smem + OFF_IDX);

    const int tid  = threadIdx.x;
    const int lane = tid & 31;
    const int wave = tid >> 5;
    const int v0   = blockIdx.x * VT;

    // --- stage conv_w as bf16 [64][PADK] ---
    for (int i = tid; i < COUTc * CK; i += 256) {
        int o = i / CK, cj = i - o * CK;
        sW[o * PADK + cj] = f2bf(conv_w[i]);
    }
    // --- stage itp_mat tile [VT][81] ---
    for (int i = tid; i < VT * 81; i += 256) {
        int vl = i / 81, e = i - vl * 81;
        int v = v0 + vl;
        sMat[i] = (v < Vc) ? itp_mat[(size_t)v * 81 + e] : 0.0f;
    }
    // --- stage index tile [VT][9] ---
    for (int i = tid; i < VT * Kc; i += 256) {
        int vl = i / Kc, k = i - vl * Kc;
        int v = v0 + vl;
        sIdx[i] = (v < Vc) ? index[(size_t)v * Kc + k] : 0;
    }
    __syncthreads();

    // --- Phase A: gather + interp -> bf16 ITP[col = b*16+vl][c*9+j] ---
    for (int vv = 0; vv < 2; ++vv) {
        const int vl = wave * 2 + vv;
        const int v  = v0 + vl;
        const bool valid = (v < Vc);
        int nb[Kc];
        #pragma unroll
        for (int k = 0; k < Kc; ++k) nb[k] = sIdx[vl * Kc + k];
        const float* mat = &sMat[vl * 81];
        for (int it = 0; it < 8; ++it) {
            const int p = it * 32 + lane;   // (b,c) pair 0..255
            const int b = p >> 6;
            const int c = p & 63;
            float g[Kc];
            #pragma unroll
            for (int k = 0; k < Kc; ++k) {
                g[k] = valid ? (useT ? xT[(size_t)nb[k] * 256 + p]
                                     : x[(size_t)p * Vc + nb[k]])
                             : 0.0f;
            }
            float acc[Kc];
            #pragma unroll
            for (int j = 0; j < Kc; ++j) acc[j] = 0.0f;
            #pragma unroll
            for (int k = 0; k < Kc; ++k) {
                const float gk = g[k];
                #pragma unroll
                for (int j = 0; j < Kc; ++j) acc[j] += gk * mat[k * Kc + j];
            }
            unsigned short* dst = &sITP[(b * VT + vl) * PADK + c * Kc];
            #pragma unroll
            for (int j = 0; j < Kc; ++j) dst[j] = f2bf(acc[j]);
        }
    }
    __syncthreads();

    // --- Phase B: OUT[64x64] = W[64x576]_bf16 @ ITP[576x64]_bf16, f32 acc ---
    const int mi   = wave & 3;        // M tile (rows of W)
    const int njA  = wave >> 2;       // N tiles njA and njA+2 (each == batch id)
    const int half = lane >> 4;       // 0: lanes 0-15, 1: lanes 16-31
    const int l16  = lane & 15;

    v8f acc0 = {}; v8f acc1 = {};
    const unsigned short* aRow  = &sW  [(mi * 16 + l16) * PADK];
    const unsigned short* bCol0 = &sITP[(njA * 16 + l16) * PADK];
    const unsigned short* bCol1 = &sITP[((njA + 2) * 16 + l16) * PADK];

    #pragma unroll 2
    for (int ks = 0; ks < NKSTEP; ++ks) {
        const int kbase = ks * 32;
        // A fragment: contiguous K order (lanes 0-15: K 0-15; 16-31: K 16-31), matches B
        const int ka = kbase + (half ? 16 : 0);
        v8bf a_lo = *(const v8bf*)(aRow + ka);
        v8bf a_hi = *(const v8bf*)(aRow + ka + 8);
        v16bf a;
        #pragma unroll
        for (int i = 0; i < 8; ++i) { a[i] = a_lo[i]; a[8 + i] = a_hi[i]; }
        // B fragments: contiguous K order, same as A (lanes 0-15: K 0-15; 16-31: K 16-31)
        const int kb = kbase + (half ? 16 : 0);
        v8bf b0l = *(const v8bf*)(bCol0 + kb);
        v8bf b0h = *(const v8bf*)(bCol0 + kb + 8);
        v16bf b0;
        #pragma unroll
        for (int i = 0; i < 8; ++i) { b0[i] = b0l[i]; b0[8 + i] = b0h[i]; }
        v8bf b1l = *(const v8bf*)(bCol1 + kb);
        v8bf b1h = *(const v8bf*)(bCol1 + kb + 8);
        v16bf b1;
        #pragma unroll
        for (int i = 0; i < 8; ++i) { b1[i] = b1l[i]; b1[8 + i] = b1h[i]; }

        acc0 = __builtin_amdgcn_wmma_f32_16x16x32_bf16(false, a, false, b0,
                                                       (short)0, acc0, false, false);
        acc1 = __builtin_amdgcn_wmma_f32_16x16x32_bf16(false, a, false, b1,
                                                       (short)0, acc1, false, false);
    }

    // --- store: D layout: VGPR r -> M = r (+8 for upper half), N = lane&15 ---
    const int v = v0 + l16;
    if (v < Vc) {
        #pragma unroll
        for (int r = 0; r < 8; ++r) {
            const int m = mi * 16 + r + (half ? 8 : 0);
            const float bias = conv_b[m];
            out[((size_t)(njA * COUTc + m)) * Vc + v]       = acc0[r] + bias;
            out[((size_t)((njA + 2) * COUTc + m)) * Vc + v] = acc1[r] + bias;
        }
    }
}

extern "C" void kernel_launch(void* const* d_in, const int* in_sizes, int n_in,
                              void* d_out, int out_size, void* d_ws, size_t ws_size,
                              hipStream_t stream) {
    const float* x       = (const float*)d_in[0];
    const int*   index   = (const int*)d_in[1];
    const float* itp_mat = (const float*)d_in[2];
    const float* conv_w  = (const float*)d_in[3];
    const float* conv_b  = (const float*)d_in[4];
    float* out = (float*)d_out;

    const size_t xT_bytes = (size_t)Vc * 256 * sizeof(float);
    const int useT = (d_ws != nullptr && ws_size >= xT_bytes) ? 1 : 0;
    float* xT = (float*)d_ws;

    (void)hipFuncSetAttribute((const void*)sphere_conv_fused_kernel,
                              hipFuncAttributeMaxDynamicSharedMemorySize,
                              SMEM_BYTES);

    if (useT) {
        dim3 tg((Vc + 63) / 64, 4);
        hipLaunchKernelGGL(transpose_x_kernel, tg, dim3(256), 0, stream, x, xT);
    }
    const int blocks = (Vc + VT - 1) / VT;
    hipLaunchKernelGGL(sphere_conv_fused_kernel, dim3(blocks), dim3(256),
                       SMEM_BYTES, stream,
                       x, xT, useT, index, itp_mat, conv_w, conv_b, out);
}
